// FullFastSTU_18863496364133
// MI455X (gfx1250) — compile-verified, matrix operand order checked by reading, not run on hardware
//
#include <hip/hip_runtime.h>
#include <hip/hip_bf16.h>

typedef _Float16 v16h __attribute__((ext_vector_type(16)));
typedef _Float16 v8h  __attribute__((ext_vector_type(8)));
typedef float    v8f  __attribute__((ext_vector_type(8)));

#define DDIM   768
#define TSEQ   1024
#define BSZ    2
#define SDIM   128
#define HALFN  24
#define BPRIME (BSZ * DDIM)   /* 1536 */
#define MROWS  (BSZ * TSEQ)   /* 2048 */

/* ---------------- elementwise converts ---------------- */

__global__ __launch_bounds__(256) void k_cvt_x(const float* __restrict__ x,
                                               _Float16* __restrict__ xh, int n) {
  int i = blockIdx.x * 256 + threadIdx.x;
  if (i < n) xh[i] = (_Float16)x[i];
}

/* MhT[n*768+k] = f16(M_inputs[k*768+n])  (pre-transpose so B frags are K-contiguous) */
__global__ __launch_bounds__(256) void k_cvt_mt(const double* __restrict__ M,
                                                _Float16* __restrict__ MhT) {
  int i = blockIdx.x * 256 + threadIdx.x;   // i = n*768 + k
  int nn = i / DDIM, k = i % DDIM;
  MhT[i] = (_Float16)(float)M[(size_t)k * DDIM + nn];
}

/* W[s*768+d] = sum_j (BC[s,j]+BC[s,j+24]) * Mf[j,d] ; also Af = f32(A) */
__global__ __launch_bounds__(256) void k_w(const double* __restrict__ Mf,
                                           const double* __restrict__ A,
                                           const double* __restrict__ BC,
                                           float* __restrict__ W,
                                           float* __restrict__ Af) {
  int i = blockIdx.x * 256 + threadIdx.x;   // i = s*768 + d
  int s = i / DDIM, d = i % DDIM;
  double acc = 0.0;
#pragma unroll
  for (int j = 0; j < HALFN; ++j)
    acc += (BC[s * 2 * HALFN + j] + BC[s * 2 * HALFN + j + HALFN]) * Mf[(size_t)j * DDIM + d];
  W[i] = (float)acc;
  if (i < SDIM) Af[i] = (float)A[i];
}

/* ---------------- async-copy helpers (CDNA5 ASYNCcnt path) ---------------- */

/* LDS byte offset = low 32 bits of the flat address */
__device__ __forceinline__ uint32_t lds_off(const void* p) {
  return (uint32_t)(uintptr_t)p;
}

/* GLOBAL_LOAD_ASYNC_TO_LDS_B128: copies 16 B from global vaddr to the LDS offset in vdst; tracked by ASYNCcnt */
__device__ __forceinline__ void async_ld_b128(uint32_t lds, const _Float16* g) {
  asm volatile("global_load_async_to_lds_b128 %0, %1, off"
               :: "v"(lds), "v"(g) : "memory");
}

__device__ __forceinline__ void wait_asynccnt0() {
  asm volatile("s_wait_asynccnt 0x0" ::: "memory");
}

/* ---------------- WMMA GEMM: xt[t*1536 + b*768 + d] = (x @ M_inputs) ----------------
   Block = 256 thr (8 waves). Block tile 128(M) x 64(N); wave w owns rows [w*16,w*16+16),
   4 N-subtiles of 16. Double-buffered LDS panels filled by async global->LDS DMA so the
   fill of K-step i+1 overlaps the WMMAs of step i; one barrier per K-step. */
__global__ __launch_bounds__(256) void k_gemm(const _Float16* __restrict__ xh,
                                              const _Float16* __restrict__ MhT,
                                              float* __restrict__ xt) {
  __shared__ _Float16 As[2][128 * 32];   // As[buf][row*32 + k]
  __shared__ _Float16 Bs[2][64 * 32];    // Bs[buf][col*32 + k] (pre-transposed vs K)

  const int tid  = threadIdx.x;
  const int wv   = tid >> 5;
  const int lane = tid & 31;
  const int hi   = lane >> 4;         // lane half selects K sub-pattern
  const int r    = lane & 15;         // row (A) / col (B/C) within tile
  const int kb   = hi << 3;           // 0 or 8
  const int m_blk = blockIdx.x * 128;
  const int n_blk = blockIdx.y * 64;

  // per-thread staging: two A chunks + one B chunk, 8 halves (16B) each
  const int rowA0 = tid >> 2;                // 0..63
  const int rowA1 = rowA0 + 64;              // 64..127
  const int kc    = (tid & 3) << 3;          // 0,8,16,24
  const int colB  = tid >> 2;                // 0..63

  const _Float16* gA0 = xh  + (size_t)(m_blk + rowA0) * DDIM + kc;
  const _Float16* gA1 = xh  + (size_t)(m_blk + rowA1) * DDIM + kc;
  const _Float16* gB  = MhT + (size_t)(n_blk + colB ) * DDIM + kc;

  const uint32_t lA0 = lds_off(&As[0][rowA0 * 32 + kc]);
  const uint32_t lA1 = lds_off(&As[0][rowA1 * 32 + kc]);
  const uint32_t lB  = lds_off(&Bs[0][colB  * 32 + kc]);
  const uint32_t strA = 128 * 32 * sizeof(_Float16);   // 8192 B between A buffers
  const uint32_t strB = 64 * 32 * sizeof(_Float16);    // 4096 B between B buffers

  // prologue: fill buffer 0 (K-step 0)
  async_ld_b128(lA0, gA0);
  async_ld_b128(lA1, gA1);
  async_ld_b128(lB , gB );

  v8f acc[4] = {};

  for (int it = 0; it < DDIM / 32; ++it) {
    const int cur = it & 1;
    wait_asynccnt0();      // my fills for buf[cur] are done
    __syncthreads();       // everyone's fills done; everyone's reads of buf[cur^1] done

    if (it + 1 < DDIM / 32) {            // launch DMA for next K-step into other buffer
      const int k0n = (it + 1) * 32;
      const uint32_t nb = (uint32_t)(cur ^ 1);
      async_ld_b128(lA0 + nb * strA, gA0 + k0n);
      async_ld_b128(lA1 + nb * strA, gA1 + k0n);
      async_ld_b128(lB  + nb * strB, gB  + k0n);
    }

    const _Float16* Ab = As[cur];
    const _Float16* Bb = Bs[cur];

    // A fragment: row = wv*16 + r; K halves = {kb..kb+7, kb+16..kb+23}
    v8h alo = *(const v8h*)(Ab + (wv * 16 + r) * 32 + kb);
    v8h ahi = *(const v8h*)(Ab + (wv * 16 + r) * 32 + kb + 16);
    v16h a = __builtin_shufflevector(alo, ahi, 0,1,2,3,4,5,6,7,8,9,10,11,12,13,14,15);

#pragma unroll
    for (int j = 0; j < 4; ++j) {
      v8h blo = *(const v8h*)(Bb + (j * 16 + r) * 32 + kb);
      v8h bhi = *(const v8h*)(Bb + (j * 16 + r) * 32 + kb + 16);
      v16h b = __builtin_shufflevector(blo, bhi, 0,1,2,3,4,5,6,7,8,9,10,11,12,13,14,15);
      acc[j] = __builtin_amdgcn_wmma_f32_16x16x32_f16(
          /*neg_a=*/false, a, /*neg_b=*/false, b,
          /*c_mod=*/(short)0, acc[j], /*reuse_a=*/false, /*reuse_b=*/false);
    }
  }

  // write out time-major: VGPR rr holds M = rr + hi*8; lane r is N
#pragma unroll
  for (int j = 0; j < 4; ++j) {
    int col = n_blk + j * 16 + r;
#pragma unroll
    for (int rr = 0; rr < 8; ++rr) {
      int row = m_blk + wv * 16 + hi * 8 + rr;   // row = b*1024 + t
      int bb = row >> 10, tt = row & 1023;
      xt[(size_t)tt * BPRIME + bb * DDIM + col] = acc[j][rr];
    }
  }
}

/* ---------------- diagonal linear scan ----------------
   6144 threads; 4 adjacent lanes share one channel (32 states each in VGPRs). */
__global__ __launch_bounds__(256) void k_scan(const float* __restrict__ xt,
                                              const float* __restrict__ W,
                                              const float* __restrict__ Af,
                                              __hip_bfloat16* __restrict__ out) {
  int g = blockIdx.x * 256 + threadIdx.x;   // 0..6143
  int c = g >> 2;                           // channel b' = b*768 + d
  int p = g & 3;                            // state quarter
  int b = c / DDIM, d = c % DDIM;

  float a[32], w[32], h[32];
#pragma unroll
  for (int i = 0; i < 32; ++i) {
    int s = p * 32 + i;
    a[i] = Af[s];
    w[i] = W[(size_t)s * DDIM + d];
    h[i] = 0.0f;
  }

  const float* xptr = xt + c;
  __hip_bfloat16* optr = out + (size_t)b * TSEQ * DDIM + d;

  for (int t = 0; t < TSEQ; ++t) {
    float xv = xptr[(size_t)t * BPRIME];
    float y0 = 0.f, y1 = 0.f, y2 = 0.f, y3 = 0.f;
#pragma unroll
    for (int i = 0; i < 32; i += 4) {
      h[i]     = fmaf(h[i],     a[i],     xv); y0 = fmaf(h[i],     w[i],     y0);
      h[i + 1] = fmaf(h[i + 1], a[i + 1], xv); y1 = fmaf(h[i + 1], w[i + 1], y1);
      h[i + 2] = fmaf(h[i + 2], a[i + 2], xv); y2 = fmaf(h[i + 2], w[i + 2], y2);
      h[i + 3] = fmaf(h[i + 3], a[i + 3], xv); y3 = fmaf(h[i + 3], w[i + 3], y3);
    }
    float y = (y0 + y1) + (y2 + y3);
    y += __shfl_xor(y, 1, 32);   // combine the 4-lane state partition
    y += __shfl_xor(y, 2, 32);
    if (p == 0) optr[(size_t)t * DDIM] = __float2bfloat16(y);
  }
}

/* ---------------- launch ---------------- */
extern "C" void kernel_launch(void* const* d_in, const int* in_sizes, int n_in,
                              void* d_out, int out_size, void* d_ws, size_t ws_size,
                              hipStream_t stream) {
  (void)in_sizes; (void)n_in; (void)out_size; (void)ws_size;
  const float*  x   = (const float*) d_in[0];
  const double* Mi  = (const double*)d_in[1];
  const double* Mf  = (const double*)d_in[2];
  const double* A   = (const double*)d_in[3];
  const double* BC  = (const double*)d_in[4];
  __hip_bfloat16* out = (__hip_bfloat16*)d_out;

  char* ws = (char*)d_ws;
  float*    xt  = (float*)   (ws + 0);          // 1024*1536*4   = 6,291,456 B
  _Float16* xh  = (_Float16*)(ws + 6291456);    // 2048*768*2    = 3,145,728 B
  _Float16* MhT = (_Float16*)(ws + 9437184);    //  768*768*2    = 1,179,648 B
  float*    W   = (float*)   (ws + 10616832);   //  128*768*4    =   393,216 B
  float*    Af  = (float*)   (ws + 11010048);   //  128*4

  k_cvt_x <<<6144, 256, 0, stream>>>(x, xh, MROWS * DDIM);
  k_cvt_mt<<<2304, 256, 0, stream>>>(Mi, MhT);
  k_w     <<< 384, 256, 0, stream>>>(Mf, A, BC, W, Af);

  dim3 grid_gemm(MROWS / 128, DDIM / 64);       // 16 x 12
  k_gemm  <<<grid_gemm, 256, 0, stream>>>(xh, MhT, xt);

  k_scan  <<<24, 256, 0, stream>>>(xt, W, Af, out);
}
